// BroGNet_44418551775943
// MI455X (gfx1250) — matrix operand order repaired and checked by reading, not compiled, not run on hardware
//
#include <hip/hip_runtime.h>
#include <hip/hip_bf16.h>
#include <math.h>

typedef __attribute__((ext_vector_type(16))) __bf16 v16bf;
typedef __attribute__((ext_vector_type(8)))  float  v8f;

#define NNODES 50000
#define NN_PAD 50048      // 391 * 128: node rows padded so stores are unguarded
#define NEDGES 800000

__device__ __forceinline__ float softplus_f(float x) {
  // branchless select; short transcendental sequence (v_exp_f32 + v_log_f32)
  float sp = __logf(1.0f + __expf(x));
  return (x > 20.0f) ? x : sp;
}

// ---------------------------------------------------------------------------
// Fused GEMM layer: Y[M,N] = act(X[M,K] @ W[K,N] + b[N]) (+ R[M,N])
// bf16 operands staged in LDS, fp32 accumulate via v_wmma_f32_16x16x32_bf16.
// KP/NP: padded K/N. ROWS: row tile per 256-thread block (8 waves).
// Grid must cover ceil(M/ROWS) blocks; Y must have ceil(M/ROWS)*ROWS rows.
// LGUARD: guard staging loads with row < M (input tensors only).
// COLGUARD: guard stores with cn < N (only when N < NP).
// Epilogue row loops are always straight-line (no per-row branching).
// ---------------------------------------------------------------------------
template<int KP, int NP, int ROWS, int ACT, int HASRES, int LGUARD, int COLGUARD>
__global__ __launch_bounds__(256)
void gemm_layer(const float* __restrict__ X, const float* __restrict__ W,
                const float* __restrict__ B, const float* __restrict__ R,
                float* __restrict__ Y, int M, int K, int N)
{
  __shared__ __bf16 sX[ROWS * KP];
  __shared__ __bf16 sW[KP * NP];
  __shared__ float  sB[NP];

  const int tid  = threadIdx.x;
  const int lane = tid & 31;
  const int wave = tid >> 5;
  const int row0 = blockIdx.x * ROWS;

  // ---- Stage activation tile as bf16 (float4 global loads; K % 4 == 0) ----
  {
    const int kc = K >> 2;                 // float4 chunks per row
    for (int c = tid; c < ROWS * kc; c += 256) {
      int rr = c / kc, c4 = c - rr * kc;
      int gr = row0 + rr, cc = c4 << 2;
      float4 v;
      if (!LGUARD || gr < M) v = *(const float4*)(X + (size_t)gr * K + cc);
      else                   v = make_float4(0.f, 0.f, 0.f, 0.f);
      __bf16* d = &sX[rr * KP + cc];
      d[0] = (__bf16)v.x; d[1] = (__bf16)v.y; d[2] = (__bf16)v.z; d[3] = (__bf16)v.w;
    }
    if (KP > K) {                          // zero the K padding
      for (int c = tid; c < ROWS * (KP - K); c += 256) {
        int rr = c / (KP - K), cc = K + (c - rr * (KP - K));
        sX[rr * KP + cc] = (__bf16)0.0f;
      }
    }
  }
  // ---- Stage weights as bf16 ----
  if ((N & 3) == 0) {
    for (int c = tid; c < (K * N) >> 2; c += 256) {
      int flat = c << 2;
      int kk = flat / N, nn = flat - kk * N;
      float4 v = *(const float4*)(W + flat);
      __bf16* d = &sW[kk * NP + nn];
      d[0] = (__bf16)v.x; d[1] = (__bf16)v.y; d[2] = (__bf16)v.z; d[3] = (__bf16)v.w;
    }
  } else {
    for (int idx = tid; idx < KP * NP; idx += 256) {
      int kk = idx / NP, nn = idx - kk * NP;
      float v = (kk < K && nn < N) ? W[(size_t)kk * N + nn] : 0.0f;
      sW[idx] = (__bf16)v;
    }
  }
  if (KP > K && (N & 3) == 0) {            // zero weight K padding (vec path)
    for (int idx = tid; idx < (KP - K) * NP; idx += 256)
      sW[K * NP + idx] = (__bf16)0.0f;
  }
  for (int idx = tid; idx < NP; idx += 256)
    sB[idx] = (idx < N) ? B[idx] : 0.0f;

  // Prefetch next row tile while we compute (global_prefetch_b8)
  if (row0 + ROWS < M && tid < 32)
    __builtin_prefetch((const char*)(X + (size_t)(row0 + ROWS) * K) + tid * 256, 0, 1);

  __syncthreads();

  constexpr int SLABS     = ROWS / 16;     // 16-row slabs per block
  constexpr int NT_STRIDE = 8 / SLABS;     // waves sharing a slab stride N-tiles
  constexpr int KT        = KP / 32;
  const int slab = wave % SLABS;
  // A-fragment per CDNA5 16-bit 16x32 layout:
  // lane l: row M=(l&15); K-base=(l>>4)*8 for elems 0..7, +16 for elems 8..15
  const int am  = slab * 16 + (lane & 15);
  const int akb = (lane >> 4) << 3;

  // Preload ALL A-fragments for this wave's slab; reuse across N-tiles.
  v16bf afrag[KT];
  #pragma unroll
  for (int kt = 0; kt < KT; ++kt) {
    const __bf16* ax = &sX[am * KP + kt * 32 + akb];
    #pragma unroll
    for (int j = 0; j < 4; ++j) { afrag[kt][2*j]   = ax[2*j];        afrag[kt][2*j+1] = ax[2*j+1]; }
    #pragma unroll
    for (int j = 0; j < 4; ++j) { afrag[kt][8+2*j] = ax[16 + 2*j];   afrag[kt][9+2*j] = ax[16+2*j+1]; }
  }

  for (int nt = wave / SLABS; nt < NP / 16; nt += NT_STRIDE) {
    v8f acc = {0.f, 0.f, 0.f, 0.f, 0.f, 0.f, 0.f, 0.f};
    #pragma unroll
    for (int kt = 0; kt < KT; ++kt) {
      // B-fragment: 32x16 bf16, lane = N column (l&15); K same as A: (l>>4)*8 + e for e<8, +16 for e>=8
      v16bf b;
      const __bf16* bx = &sW[(kt * 32 + ((lane >> 4) << 3)) * NP + nt * 16 + (lane & 15)];
      #pragma unroll
      for (int e2 = 0; e2 < 16; ++e2) b[e2] = bx[((e2 & 7) + ((e2 >> 3) << 4)) * NP];
      acc = __builtin_amdgcn_wmma_f32_16x16x32_bf16(
          false, afrag[kt], false, b, (short)0, acc, false, false);
    }
    // C/D layout: VGPR r, lanes 0-15 -> M=r, N=lane; lanes 16-31 -> M=8+r
    const int cn   = nt * 16 + (lane & 15);
    const int mb   = slab * 16 + ((lane >> 4) << 3);
    const float bv = sB[cn];
    if (!COLGUARD || cn < N) {             // single uniform-per-lane predicate
      #pragma unroll
      for (int r2 = 0; r2 < 8; ++r2) {     // straight-line: no row guards
        size_t o = (size_t)(row0 + mb + r2) * N + cn;
        float y = acc[r2] + bv;
        if (ACT)    y = softplus_f(y);
        if (HASRES) y += R[o];
        Y[o] = y;
      }
    }
  }
}

// ---------------------------------------------------------------------------
// Elementwise / gather / scatter kernels (float4; rows are 16B-aligned)
// ---------------------------------------------------------------------------
__global__ void edge_prod_k(const float* __restrict__ h, const int* __restrict__ ei,
                            float* __restrict__ out)
{
  long long stride = (long long)gridDim.x * blockDim.x;
  long long total  = (long long)NEDGES * 24;           // 96/4 float4 per edge
  for (long long idx = (long long)blockIdx.x * blockDim.x + threadIdx.x;
       idx < total; idx += stride) {
    int i = (int)(idx / 24), c = (int)(idx - (long long)i * 24);
    int s = ei[i], r = ei[NEDGES + i];
    float4 a = *(const float4*)(h + (size_t)s * 96 + 4 * c);
    float4 b = *(const float4*)(h + (size_t)r * 96 + 4 * c);
    float4 o = make_float4(a.x * b.x, a.y * b.y, a.z * b.z, a.w * b.w);
    *(float4*)(out + (size_t)i * 96 + 4 * c) = o;
  }
}

__global__ void concat_k(const float* __restrict__ h, const float* __restrict__ e,
                         const int* __restrict__ ei, float* __restrict__ out)
{
  long long stride = (long long)gridDim.x * blockDim.x;
  long long total  = (long long)NEDGES * 48;           // 192/4 float4 per edge
  for (long long idx = (long long)blockIdx.x * blockDim.x + threadIdx.x;
       idx < total; idx += stride) {
    int i = (int)(idx / 48), c = (int)(idx - (long long)i * 48);
    float4 v;
    if (c < 24) {
      int r = ei[NEDGES + i];
      v = *(const float4*)(h + (size_t)r * 96 + 4 * c);
    } else {
      v = *(const float4*)(e + (size_t)i * 96 + 4 * (c - 24));
    }
    *(float4*)(out + (size_t)i * 192 + 4 * c) = v;
  }
}

__global__ void scatter_h_k(const float* __restrict__ msg, const int* __restrict__ ei,
                            float* __restrict__ h)
{
  long long stride = (long long)gridDim.x * blockDim.x;
  long long total  = (long long)NEDGES * 24;
  for (long long idx = (long long)blockIdx.x * blockDim.x + threadIdx.x;
       idx < total; idx += stride) {
    int i = (int)(idx / 24), c = (int)(idx - (long long)i * 24);
    int r = ei[NEDGES + i];
    float4 m = *(const float4*)(msg + (size_t)i * 96 + 4 * c);
    float* dst = h + (size_t)r * 96 + 4 * c;
    atomicAdd(dst + 0, m.x);
    atomicAdd(dst + 1, m.y);
    atomicAdd(dst + 2, m.z);
    atomicAdd(dst + 3, m.w);
  }
}

__global__ void force_k(const float* __restrict__ fij, const int* __restrict__ ei,
                        float* __restrict__ ai)
{
  long long stride = (long long)gridDim.x * blockDim.x;
  long long total  = (long long)NEDGES * 3;
  for (long long idx = (long long)blockIdx.x * blockDim.x + threadIdx.x;
       idx < total; idx += stride) {
    int i = (int)(idx / 3), c = (int)(idx - (long long)i * 3);
    float v = fij[idx];
    int s = ei[i], r = ei[NEDGES + i];
    atomicAdd(&ai[(size_t)r * 3 + c],  v);
    atomicAdd(&ai[(size_t)s * 3 + c], -v);
  }
}

__global__ void zero_k(float* __restrict__ p, long long n)
{
  long long stride = (long long)gridDim.x * blockDim.x;
  for (long long i = (long long)blockIdx.x * blockDim.x + threadIdx.x;
       i < n; i += stride) p[i] = 0.0f;
}

// mlp2: 5 -> 5 -> 5 -> 1, softplus after EVERY layer (final_act=True)
__global__ void mlp2_k(const float* __restrict__ t,
                       const float* __restrict__ W0, const float* __restrict__ b0,
                       const float* __restrict__ W1, const float* __restrict__ b1,
                       const float* __restrict__ W2, const float* __restrict__ b2,
                       float* __restrict__ gamma, int n)
{
  int i = blockIdx.x * blockDim.x + threadIdx.x;
  if (i >= n) return;
  float in[5], a[5], b[5];
  #pragma unroll
  for (int j = 0; j < 5; ++j) in[j] = t[(size_t)i * 5 + j];
  #pragma unroll
  for (int o = 0; o < 5; ++o) {
    float acc = b0[o];
    #pragma unroll
    for (int j = 0; j < 5; ++j) acc += in[j] * W0[j * 5 + o];
    a[o] = softplus_f(acc);
  }
  #pragma unroll
  for (int o = 0; o < 5; ++o) {
    float acc = b1[o];
    #pragma unroll
    for (int j = 0; j < 5; ++j) acc += a[j] * W1[j * 5 + o];
    b[o] = softplus_f(acc);
  }
  float acc = b2[0];
  #pragma unroll
  for (int j = 0; j < 5; ++j) acc += b[j] * W2[j];
  gamma[i] = softplus_f(acc);
}

// ---------------------------------------------------------------------------
static inline int ew_blocks(long long total) {
  long long nb = (total + 255) / 256;
  return (int)(nb > 131072 ? 131072 : nb);
}

#define GEMM(KP, NP, RWS, ACT, RES, LG, CG, X, W, B, R, Y, M, K, Nn)           \
  gemm_layer<KP, NP, RWS, ACT, RES, LG, CG>                                    \
      <<<dim3(((M) + (RWS) - 1) / (RWS)), dim3(256), 0, stream>>>(             \
          X, W, B, R, Y, M, K, Nn)

extern "C" void kernel_launch(void* const* d_in, const int* in_sizes, int n_in,
                              void* d_out, int out_size, void* d_ws, size_t ws_size,
                              hipStream_t stream) {
  const float* x         = (const float*)d_in[0];
  const float* edge_attr = (const float*)d_in[1];
  const float* type_ohe  = (const float*)d_in[2];
  const float* faW0 = (const float*)d_in[3],  *fab0 = (const float*)d_in[4];
  const float* faW1 = (const float*)d_in[5],  *fab1 = (const float*)d_in[6];
  const float* fbW0 = (const float*)d_in[7],  *fbb0 = (const float*)d_in[8];
  const float* fbW1 = (const float*)d_in[9],  *fbb1 = (const float*)d_in[10];
  const float* feW0 = (const float*)d_in[11], *feb0 = (const float*)d_in[12];
  const float* feW1 = (const float*)d_in[13], *feb1 = (const float*)d_in[14];
  const float* feW2 = (const float*)d_in[15], *feb2 = (const float*)d_in[16];
  const float* fvW0 = (const float*)d_in[17], *fvb0 = (const float*)d_in[18];
  const float* fvW1 = (const float*)d_in[19], *fvb1 = (const float*)d_in[20];
  const float* fvW2 = (const float*)d_in[21], *fvb2 = (const float*)d_in[22];
  const float* m1W0 = (const float*)d_in[23], *m1b0 = (const float*)d_in[24];
  const float* m1W1 = (const float*)d_in[25], *m1b1 = (const float*)d_in[26];
  const float* m1W2 = (const float*)d_in[27], *m1b2 = (const float*)d_in[28];
  const float* m2W0 = (const float*)d_in[29], *m2b0 = (const float*)d_in[30];
  const float* m2W1 = (const float*)d_in[31], *m2b1 = (const float*)d_in[32];
  const float* m2W2 = (const float*)d_in[33], *m2b2 = (const float*)d_in[34];
  const int*   ei   = (const int*)d_in[35];
  const float* NORES = nullptr;

  float* out = (float*)d_out;        // [ai: 50000*3 | gamma: 50000]
  float* ws  = (float*)d_ws;
  float* h   = ws;                                   // NN_PAD*96 (padded rows)
  float* e   = h  + (size_t)NN_PAD * 96;             // E*96
  float* t0  = e  + (size_t)NEDGES * 96;             // E*192
  float* t1  = t0 + (size_t)NEDGES * 192;            // E*128 (>= NN_PAD*96)

  // Node encoder: h = L1(softplus(L0(x)))
  // Stores padded to NN_PAD rows (grid 391*128) -> epilogues unguarded;
  // only the load of `x` is row-guarded. Pad rows are garbage, never gathered.
  GEMM(32,  96, 128, 1, 0, 1, 0, x,  faW0, fab0, NORES, t1, NNODES, 16, 96);
  GEMM(96,  96, 128, 0, 0, 0, 0, t1, faW1, fab1, NORES, h,  NN_PAD, 96, 96);
  // Edge encoder: e = L1(softplus(L0(edge_attr)))  (M=800000, fully unguarded)
  GEMM(32,  96, 128, 1, 0, 0, 0, edge_attr, fbW0, fbb0, NORES, t0, NEDGES, 8,  96);
  GEMM(96,  96, 128, 0, 0, 0, 0, t0,        fbW1, fbb1, NORES, e,  NEDGES, 96, 96);

  for (int step = 0; step < 4; ++step) {
    // EdgeModel: e = fe(h[s]*h[r]) + e
    edge_prod_k<<<ew_blocks((long long)NEDGES * 24), 256, 0, stream>>>(h, ei, t0);
    GEMM(96,  128, 128, 1, 0, 0, 0, t0, feW0, feb0, NORES, t1, NEDGES, 96,  128);
    GEMM(128, 128, 64,  1, 0, 0, 0, t1, feW1, feb1, NORES, t0, NEDGES, 128, 128);
    GEMM(128, 96,  128, 0, 1, 0, 0, t0, feW2, feb2, e,     e,  NEDGES, 128, 96);
    // NodeModel: h += segment_sum(fv([h[r], e]), r)
    concat_k<<<ew_blocks((long long)NEDGES * 48), 256, 0, stream>>>(h, e, ei, t0);
    GEMM(192, 128, 32,  1, 0, 0, 0, t0, fvW0, fvb0, NORES, t1, NEDGES, 192, 128);
    GEMM(128, 128, 64,  1, 0, 0, 0, t1, fvW1, fvb1, NORES, t0, NEDGES, 128, 128);
    GEMM(128, 96,  128, 0, 0, 0, 0, t0, fvW2, fvb2, NORES, t1, NEDGES, 128, 96);
    scatter_h_k<<<ew_blocks((long long)NEDGES * 24), 256, 0, stream>>>(t1, ei, h);
  }

  // mlp1: fij = mlp1(e)  (E x 3); last layer: column guard only (N=3 < NP=16)
  GEMM(96,  128, 128, 1, 0, 0, 0, e,  m1W0, m1b0, NORES, t1, NEDGES, 96,  128);
  GEMM(128, 128, 64,  1, 0, 0, 0, t1, m1W1, m1b1, NORES, t0, NEDGES, 128, 128);
  GEMM(128, 16,  128, 0, 0, 0, 1, t0, m1W2, m1b2, NORES, t1, NEDGES, 128, 3);

  // ai = seg_sum(fij, r) - seg_sum(fij, s)
  zero_k<<<ew_blocks((long long)NNODES * 3), 256, 0, stream>>>(out, (long long)NNODES * 3);
  force_k<<<ew_blocks((long long)NEDGES * 3), 256, 0, stream>>>(t1, ei, out);

  // gamma = mlp2(type_ohe) with softplus after every layer
  mlp2_k<<<(NNODES + 255) / 256, 256, 0, stream>>>(type_ohe, m2W0, m2b0, m2W1, m2b1,
                                                   m2W2, m2b2, out + (size_t)NNODES * 3,
                                                   NNODES);
}
